// ScaledDotProductAttention_bias_63333587747600
// MI455X (gfx1250) — matrix operand order repaired and checked by reading, not compiled, not run on hardware
//
#include <hip/hip_runtime.h>

typedef __attribute__((ext_vector_type(16))) _Float16 v16h;
typedef __attribute__((ext_vector_type(8)))  _Float16 v8h;
typedef __attribute__((ext_vector_type(8)))  float    v8f;
typedef __attribute__((ext_vector_type(4)))  float    v4f;

#define D_MODEL 512
#define L_SEQ   512
#define N_HEADS 8
#define D_HEAD  64
#define BKSEQ   32                 // B*K sequences
#define M_ROWS  (BKSEQ * L_SEQ)    // 16384
#define N_HEADTOT (BKSEQ * N_HEADS)// 256 heads
#define NEG_INF_V (-1000000000.0f)
#define INV_TEMP  (0.125f)
#define ROPE_LN_OVER_HALF 0.28782313662425574f  // ln(10000)/32

// ---- WMMA fragment builders (layouts per CDNA5 ISA §7.12.2) ----

// B 32x16 f16: lane holds col = lane&15, K = (lane>=16?8:0) + e (e<8), 16 + (lane>=16?8:0) + (e-8) (e>=8);
// p -> two 8-half runs at p and p+16 (16B-aligned).
__device__ __forceinline__ v16h make_b16(const _Float16* p) {
  v8h b0 = *(const v8h*)p;
  v8h b1 = *(const v8h*)(p + 16);
  v16h b;
#pragma unroll
  for (int i = 0; i < 8; ++i) { b[i] = b0[i]; b[i + 8] = b1[i]; }
  return b;
}

// A 16x32 f16 from two contiguous 8-half runs (K=kb.. and K=kb+16..).
__device__ __forceinline__ v16h make_a_f16(const _Float16* p0, const _Float16* p1) {
  v8h x0 = *(const v8h*)p0;
  v8h x1 = *(const v8h*)p1;
  v16h a;
#pragma unroll
  for (int i = 0; i < 8; ++i) { a[i] = x0[i]; a[i + 8] = x1[i]; }
  return a;
}

// Same but converting from two 8-float f32 runs.
__device__ __forceinline__ v16h make_a_f32(const float* p0, const float* p1) {
  v4f x0 = *(const v4f*)p0;
  v4f x1 = *(const v4f*)(p0 + 4);
  v4f x2 = *(const v4f*)p1;
  v4f x3 = *(const v4f*)(p1 + 4);
  v16h a;
#pragma unroll
  for (int i = 0; i < 4; ++i) {
    a[i]      = (_Float16)x0[i];
    a[i + 4]  = (_Float16)x1[i];
    a[i + 8]  = (_Float16)x2[i];
    a[i + 12] = (_Float16)x3[i];
  }
  return a;
}

// ---------------- Kernel 0: transpose + f16-convert the weights -----------
// wt[z][n][k] = (f16) W_z[k][n]   (z = 0:Wq 1:Wk 2:Wv), 512x512 each
__global__ void __launch_bounds__(256)
prep_w_kernel(const float* __restrict__ Wq, const float* __restrict__ Wk,
              const float* __restrict__ Wv, _Float16* __restrict__ wt) {
  int idx = blockIdx.x * 256 + threadIdx.x;       // 0 .. 3*262144-1
  int z   = idx >> 18;
  int rem = idx & 262143;
  int kr  = rem >> 9;                              // row of W (input-dim)
  int n   = rem & 511;                             // col of W (output-dim)
  const float* W = (z == 0) ? Wq : ((z == 1) ? Wk : Wv);
  wt[((size_t)z << 18) + ((size_t)n << 9) + kr] = (_Float16)W[((size_t)kr << 9) + n];
}

// ---------------- Kernel 1: projection GEMM + RoPE, f16 outputs -----------
// Y = X @ W  (M=16384, N=512, K=512). z=0 -> qh (RoPE), 1 -> kh (RoPE),
// 2 -> vhT (transposed per head: [head][d][t]).
// Block: 64 rows x 128 cols; wave = 16 rows x 64 cols (one full head strip).
__global__ void __launch_bounds__(256)
proj_rope_kernel(const float* __restrict__ Q, const float* __restrict__ Kin,
                 const float* __restrict__ V, const _Float16* __restrict__ wt,
                 _Float16* __restrict__ qh, _Float16* __restrict__ kh,
                 _Float16* __restrict__ vhT) {
  const int z = blockIdx.z;
  const float* X = (z == 0) ? Q : ((z == 1) ? Kin : V);
  const _Float16* W = wt + ((size_t)z << 18);

  const int mbase = blockIdx.x * 64;
  const int nbase = blockIdx.y * 128;
  const int wid   = threadIdx.x >> 5;
  const int lane  = threadIdx.x & 31;
  const int mt    = wid & 3;            // 16-row tile within block
  const int nw    = wid >> 2;           // which 64-col strip
  const int lrow  = lane & 15;
  const bool hi   = lane >= 16;
  const int kbA   = hi ? 8 : 0;
  const int koffB = hi ? 8 : 0;
  const int arow  = mbase + mt * 16 + lrow;
  const int ncol0 = nbase + nw * 64;

  const float* xrow = X + (size_t)arow * D_MODEL;
  v8f acc[4] = {};

  for (int kk = 0; kk < D_MODEL; kk += 32) {
    __builtin_prefetch(xrow + kk + 64, 0, 0);
    v16h a = make_a_f32(xrow + kk + kbA, xrow + kk + 16 + kbA);
#pragma unroll
    for (int j = 0; j < 4; ++j) {
      const _Float16* wp = W + (size_t)(ncol0 + j * 16 + lrow) * D_MODEL + kk + koffB;
      v16h b = make_b16(wp);
      acc[j] = __builtin_amdgcn_wmma_f32_16x16x32_f16(false, a, false, b,
                                                      (short)0, acc[j], false, false);
    }
  }

  // Epilogue: RoPE (for q,k) + pack to f16 head-major buffers.
  const int head = ncol0 >> 6;  // head index within [0,8)
  _Float16* qk_dst_base = (z == 0) ? qh : kh;
  // frequency for pairs (acc0,acc2): d = lrow; (acc1,acc3): d = 16+lrow
  const float invf0 = expf(-(float)lrow * ROPE_LN_OVER_HALF);
  const float invf1 = expf(-(float)(16 + lrow) * ROPE_LN_OVER_HALF);

#pragma unroll
  for (int i = 0; i < 8; ++i) {
    const int m = mbase + mt * 16 + i + (hi ? 8 : 0);   // C row
    const int s = m >> 9;                               // sequence (b*K+k)
    const int t = m & 511;                              // position
    if (z < 2) {
      const float a0 = (float)t * invf0;
      const float a1 = (float)t * invf1;
      const float c0 = cosf(a0), s0 = sinf(a0);
      const float c1 = cosf(a1), s1 = sinf(a1);
      const float x1a = acc[0][i], x2a = acc[2][i];
      const float x1b = acc[1][i], x2b = acc[3][i];
      const float r0 = x1a * c0 - x2a * s0;
      const float r2 = x1a * s0 + x2a * c0;
      const float r1 = x1b * c1 - x2b * s1;
      const float r3 = x1b * s1 + x2b * c1;
      _Float16* dst = qk_dst_base + (((size_t)(s * N_HEADS + head) * L_SEQ + t) * D_HEAD);
      dst[0 * 16 + lrow] = (_Float16)r0;
      dst[1 * 16 + lrow] = (_Float16)r1;
      dst[2 * 16 + lrow] = (_Float16)r2;
      dst[3 * 16 + lrow] = (_Float16)r3;
    } else {
      _Float16* dst = vhT + ((size_t)(s * N_HEADS + head) * D_HEAD) * L_SEQ + t;
#pragma unroll
      for (int j = 0; j < 4; ++j)
        dst[(size_t)(j * 16 + lrow) * L_SEQ] = (_Float16)acc[j][i];
    }
  }
}

// ---------------- Kernel 2: S = qh @ khT / T + bias (+mask) ---------------
// Per head: M=512, N=512, K=64. Block = 64x128 tile; grid = (32, 256).
__global__ void __launch_bounds__(256)
qk_bias_kernel(const _Float16* __restrict__ qh, const _Float16* __restrict__ kh,
               const unsigned char* __restrict__ mask,
               const float* __restrict__ alpha, const float* __restrict__ beta,
               float* __restrict__ attn) {
  const int head  = blockIdx.y;                 // s*8 + h
  const int mbase = (blockIdx.x & 7) * 64;
  const int nbase = (blockIdx.x >> 3) * 128;
  const int wid   = threadIdx.x >> 5;
  const int lane  = threadIdx.x & 31;
  const int mt    = wid & 3;
  const int nw    = wid >> 2;
  const int lrow  = lane & 15;
  const bool hi   = lane >= 16;
  const int kbA   = hi ? 8 : 0;
  const int koffB = hi ? 8 : 0;
  const int arow  = mbase + mt * 16 + lrow;
  const int ncol0 = nbase + nw * 64;

  const _Float16* qhead = qh + (size_t)head * L_SEQ * D_HEAD;
  const _Float16* khead = kh + (size_t)head * L_SEQ * D_HEAD;

  v8f acc[4] = {};
#pragma unroll
  for (int kk = 0; kk < D_HEAD; kk += 32) {
    const _Float16* ap = qhead + (size_t)arow * D_HEAD + kk;
    v16h a = make_a_f16(ap + kbA, ap + 16 + kbA);
#pragma unroll
    for (int j = 0; j < 4; ++j) {
      const _Float16* bp = khead + (size_t)(ncol0 + j * 16 + lrow) * D_HEAD + kk + koffB;
      v16h b = make_b16(bp);
      acc[j] = __builtin_amdgcn_wmma_f32_16x16x32_f16(false, a, false, b,
                                                      (short)0, acc[j], false, false);
    }
  }

  const int h  = head & 7;
  const int s  = head >> 3;
  const int bb = s >> 3;                        // batch index (K=8)
  const float al = alpha[h], be = beta[h];
  float* Shead = attn + (size_t)head * L_SEQ * L_SEQ;
  const unsigned char* mrow = mask + ((size_t)bb * N_HEADS) * L_SEQ * L_SEQ; // mask[b][0]

#pragma unroll
  for (int i = 0; i < 8; ++i) {
    const int l = mbase + mt * 16 + i + (hi ? 8 : 0);
#pragma unroll
    for (int j = 0; j < 4; ++j) {
      const int t = ncol0 + j * 16 + lrow;
      float v = acc[j][i] * INV_TEMP;
      v += (l == t) ? al : be;
      if (mrow[(size_t)l * L_SEQ + t]) v = NEG_INF_V;
      Shead[(size_t)l * L_SEQ + t] = v;
    }
  }
}

// ---------------- Kernel 3: row softmax in place ---------------------------
__global__ void __launch_bounds__(128) softmax_kernel(float* __restrict__ attn) {
  float* p = attn + (size_t)blockIdx.x * L_SEQ;
  const int tid = threadIdx.x;
  v4f v = *(const v4f*)(p + tid * 4);
  float m = fmaxf(fmaxf(v[0], v[1]), fmaxf(v[2], v[3]));
#pragma unroll
  for (int off = 16; off > 0; off >>= 1) m = fmaxf(m, __shfl_xor(m, off, 32));
  __shared__ float red[4];
  const int w = tid >> 5;
  if ((tid & 31) == 0) red[w] = m;
  __syncthreads();
  m = fmaxf(fmaxf(red[0], red[1]), fmaxf(red[2], red[3]));
  __syncthreads();
  const float e0 = expf(v[0] - m), e1 = expf(v[1] - m);
  const float e2 = expf(v[2] - m), e3 = expf(v[3] - m);
  float sum = e0 + e1 + e2 + e3;
#pragma unroll
  for (int off = 16; off > 0; off >>= 1) sum += __shfl_xor(sum, off, 32);
  if ((tid & 31) == 0) red[w] = sum;
  __syncthreads();
  const float inv = 1.0f / (red[0] + red[1] + red[2] + red[3]);
  v4f r; r[0] = e0 * inv; r[1] = e1 * inv; r[2] = e2 * inv; r[3] = e3 * inv;
  *(v4f*)(p + tid * 4) = r;
}

// ---------------- Kernel 4: out = attn @ vh --------------------------------
// Per head: M=512, N=64, K=512. Block = 128 rows x 64 cols; wave = 16x64.
__global__ void __launch_bounds__(256)
pv_kernel(const float* __restrict__ attn, const _Float16* __restrict__ vhT,
          float* __restrict__ out) {
  const int head  = blockIdx.y;
  const int mbase = blockIdx.x * 128;
  const int wid   = threadIdx.x >> 5;
  const int lane  = threadIdx.x & 31;
  const int lrow  = lane & 15;
  const bool hi   = lane >= 16;
  const int kbA   = hi ? 8 : 0;
  const int koffB = hi ? 8 : 0;
  const int arow  = mbase + wid * 16 + lrow;

  const float* Ahead = attn + (size_t)head * L_SEQ * L_SEQ;
  const _Float16* Vh = vhT + (size_t)head * D_HEAD * L_SEQ;
  const float* prow = Ahead + (size_t)arow * L_SEQ;

  v8f acc[4] = {};
  for (int kk = 0; kk < L_SEQ; kk += 32) {
    __builtin_prefetch(prow + kk + 64, 0, 0);
    v16h a = make_a_f32(prow + kk + kbA, prow + kk + 16 + kbA);
#pragma unroll
    for (int j = 0; j < 4; ++j) {
      const _Float16* bp = Vh + (size_t)(j * 16 + lrow) * L_SEQ + kk + koffB;
      v16h b = make_b16(bp);
      acc[j] = __builtin_amdgcn_wmma_f32_16x16x32_f16(false, a, false, b,
                                                      (short)0, acc[j], false, false);
    }
  }

  const int s = head >> 3;
  const int h = head & 7;
#pragma unroll
  for (int i = 0; i < 8; ++i) {
    const int l = mbase + wid * 16 + i + (hi ? 8 : 0);
    float* orow = out + ((size_t)s * L_SEQ + l) * (N_HEADS * D_HEAD) + h * D_HEAD;
#pragma unroll
    for (int j = 0; j < 4; ++j) orow[j * 16 + lrow] = acc[j][i];
  }
}

// ---------------------------- launcher -------------------------------------
extern "C" void kernel_launch(void* const* d_in, const int* in_sizes, int n_in,
                              void* d_out, int out_size, void* d_ws, size_t ws_size,
                              hipStream_t stream) {
  (void)in_sizes; (void)n_in; (void)out_size; (void)ws_size;
  const float* q  = (const float*)d_in[0];
  const float* k  = (const float*)d_in[1];
  const float* v  = (const float*)d_in[2];
  const unsigned char* mask = (const unsigned char*)d_in[3];  // all-false bool
  const float* Wq = (const float*)d_in[4];
  const float* Wk = (const float*)d_in[5];
  const float* Wv = (const float*)d_in[6];
  const float* alpha = (const float*)d_in[7];
  const float* beta  = (const float*)d_in[8];

  float* out  = (float*)d_out;                                   // 8,388,608 f32
  float* attn = out + (size_t)BKSEQ * L_SEQ * (N_HEADS * D_HEAD); // 67,108,864 f32

  _Float16* wt  = (_Float16*)d_ws;                       // 3 x 512 x 512
  _Float16* qh  = wt + (size_t)3 * D_MODEL * D_MODEL;    // [256 heads][512][64]
  _Float16* kh  = qh + (size_t)N_HEADTOT * L_SEQ * D_HEAD;
  _Float16* vhT = kh + (size_t)N_HEADTOT * L_SEQ * D_HEAD; // [256 heads][64][512]

  prep_w_kernel<<<3072, 256, 0, stream>>>(Wq, Wk, Wv, wt);
  proj_rope_kernel<<<dim3(M_ROWS / 64, D_MODEL / 128, 3), 256, 0, stream>>>(
      q, k, v, wt, qh, kh, vhT);
  qk_bias_kernel<<<dim3(32, N_HEADTOT), 256, 0, stream>>>(
      qh, kh, mask, alpha, beta, attn);
  softmax_kernel<<<dim3(N_HEADTOT * L_SEQ), 128, 0, stream>>>(attn);
  pv_kernel<<<dim3(L_SEQ / 128, N_HEADTOT), 256, 0, stream>>>(attn, vhT, out);
}
